// ClientModel_26259430048403
// MI455X (gfx1250) — matrix operand order repaired and checked by reading, not compiled, not run on hardware
//
#include <hip/hip_runtime.h>
#include <hip/hip_bf16.h>
#include <math.h>

// ---------------------------------------------------------------------------
// 2-layer LSTM LM forward (B=2048, T=80, H=256, EMB=8, VOCAB=80) for gfx1250.
// Persistent-WG LSTM: one workgroup (128 thr = 4 waves) owns 16 batch rows for
// all 80 timesteps of BOTH layers; recurrent GEMMs use v_wmma_f32_16x16x32_f16
// with f32 accumulation, all state (gates, h1/h2, c1/c2) in LDS.
// ---------------------------------------------------------------------------

#define BATCH   2048
#define TSTEPS  80
#define VOCAB   80
#define EMB     8
#define HSZ     256
#define G4      1024   // 4*H
#define FORGET_BIAS 1.0f

typedef _Float16 v16h __attribute__((ext_vector_type(16)));
typedef _Float16 v8h  __attribute__((ext_vector_type(8)));
typedef float    v8f  __attribute__((ext_vector_type(8)));

__device__ __forceinline__ float sigmoid_f(float x) {
    return 1.0f / (1.0f + __expf(-x));
}
__device__ __forceinline__ float tanh_f(float x) {
    float e = __expf(-2.0f * fabsf(x));
    float t = (1.0f - e) / (1.0f + e);
    return x >= 0.0f ? t : -t;
}

// ---------------------------------------------------------------------------
// Weight prep: f32 -> f16 with B-operand swizzle for v_wmma_f32_16x16x32_f16.
// B tile (32x16) lane layout: lane L holds col n = nt*16 + (L&15),
// K halves e=0..15 at k = kt*32 + (L<16?0:16) + e, stored contiguously so a
// tile load is one 32-byte (v16h) global load per lane.
// dst[((nt*KT + kt)*32 + lane)*16 + e]
// ---------------------------------------------------------------------------
__global__ void prep_weights_kernel(const float* __restrict__ W1,
                                    const float* __restrict__ W2,
                                    _Float16* __restrict__ wh1s,   // 256x1024
                                    _Float16* __restrict__ w2s)    // 512x1024
{
    int idx = blockIdx.x * blockDim.x + threadIdx.x;
    if (idx < 256 * G4) {  // W1 hidden part: rows 8..263
        int k = idx >> 10, n = idx & (G4 - 1);
        float v = W1[(k + EMB) * G4 + n];
        int kt = k >> 5, kl = k & 31, khi = kl >> 4, e = kl & 15;
        int lane = khi * 16 + (n & 15), nt = n >> 4;
        wh1s[(((nt * 8 + kt) * 32) + lane) * 16 + e] = (_Float16)v;
    }
    if (idx < 512 * G4) {  // W2 full: rows 0..255 = h1 input, 256..511 = h2
        int k = idx >> 10, n = idx & (G4 - 1);
        float v = W2[k * G4 + n];
        int kt = k >> 5, kl = k & 31, khi = kl >> 4, e = kl & 15;
        int lane = khi * 16 + (n & 15), nt = n >> 4;
        w2s[(((nt * 16 + kt) * 32) + lane) * 16 + e] = (_Float16)v;
    }
}

// ---------------------------------------------------------------------------
// Persistent fused 2-layer LSTM. Grid: 128 blocks x 128 threads.
// Block b owns batch rows [16b, 16b+16). LDS holds gates[16][1024] f32,
// x2[16][512] f16 (cols 0..255 = h1, 256..511 = h2), c1/c2[16][256] f32.
// ---------------------------------------------------------------------------
__global__ __launch_bounds__(128)
void lstm2_persistent_kernel(const int*   __restrict__ features, // [B][T]
                             const float* __restrict__ emb,      // [VOCAB][8]
                             const float* __restrict__ W1,       // [264][1024] (x-part rows 0..7)
                             const float* __restrict__ b1,       // [1024]
                             const float* __restrict__ b2,       // [1024]
                             const _Float16* __restrict__ wh1s,  // swizzled 256x1024
                             const _Float16* __restrict__ w2s,   // swizzled 512x1024
                             float* __restrict__ h2out)          // [B][256]
{
    __shared__ float    gates[16 * G4];   // 64 KB
    __shared__ _Float16 x2[16 * 512];     // 16 KB  (h1 | h2), A-operand source
    __shared__ float    c1s[16 * HSZ];    // 16 KB
    __shared__ float    c2s[16 * HSZ];    // 16 KB
    __shared__ int      feats[16];

    const int tid  = threadIdx.x;
    const int lane = tid & 31;
    const int wave = tid >> 5;          // 0..3, splits the 64 N-tiles
    const int r0   = blockIdx.x * 16;   // first batch row of this WG

    const int m    = lane & 15;         // A row / C col within tile
    const int hi   = lane >> 4;         // lane half
    const int aoff = hi ? 16 : 0;       // A operand K base per lane half
    const int crow = hi * 8;            // C/D row base per lane half

    // ---- init state ----
    for (int i = tid; i < 16 * HSZ; i += 128) { c1s[i] = 0.0f; c2s[i] = 0.0f; }
    for (int i = tid; i < 16 * 512; i += 128) x2[i] = (_Float16)0.0f;
    if (tid < 16) feats[tid] = features[(r0 + tid) * TSTEPS + 0];
    __syncthreads();

    for (int t = 0; t < TSTEPS; ++t) {
        // ---- Phase A: gates = b1 + x_t @ W1x  (K=8, VALU) ----
        {
            float xr[EMB];
            #pragma unroll 1
            for (int i = 0; i < 128; ++i) {
                int r = i >> 3;                    // uniform per iteration
                int n = ((i & 7) << 7) | tid;
                if ((i & 7) == 0) {
                    int f = feats[r];
                    #pragma unroll
                    for (int e = 0; e < EMB; ++e) xr[e] = emb[f * EMB + e];
                }
                float acc = b1[n];
                #pragma unroll
                for (int e = 0; e < EMB; ++e) acc += xr[e] * W1[e * G4 + n];
                gates[r * G4 + n] = acc;
            }
        }
        __syncthreads();

        // ---- GEMM1: gates += h1_{t-1} @ Wh1   (WMMA f16, K=256) ----
        #pragma unroll 1
        for (int nti = 0; nti < 16; ++nti) {
            int nt = wave + nti * 4;
            int ncol = nt * 16 + m;
            v8f c;
            #pragma unroll
            for (int v = 0; v < 8; ++v) c[v] = gates[(crow + v) * G4 + ncol];
            #pragma unroll
            for (int kt = 0; kt < 8; ++kt) {
                int koff = kt * 32 + aoff;
                v8h alo = *(const v8h*)&x2[m * 512 + koff];
                v8h ahi = *(const v8h*)&x2[m * 512 + koff + 8];
                v16h a = __builtin_shufflevector(alo, ahi,
                           0,1,2,3,4,5,6,7,8,9,10,11,12,13,14,15);
                v16h b = *(const v16h*)(wh1s + (((nt * 8 + kt) * 32) + lane) * 16);
                c = __builtin_amdgcn_wmma_f32_16x16x32_f16(
                        false, a, false, b, (short)0, c, false, false);
            }
            #pragma unroll
            for (int v = 0; v < 8; ++v) gates[(crow + v) * G4 + ncol] = c[v];
        }
        __syncthreads();

        // ---- Cell 1 (gate order i, j, f, o) + reinit gates with b2 ----
        #pragma unroll 1
        for (int i = 0; i < 32; ++i) {
            int r = i >> 1;
            int n = ((i & 1) << 7) | tid;
            float gi = gates[r * G4 + n];
            float gj = gates[r * G4 + n + 256];
            float gf = gates[r * G4 + n + 512];
            float go = gates[r * G4 + n + 768];
            float c  = c1s[r * HSZ + n];
            c = c * sigmoid_f(gf + FORGET_BIAS) + sigmoid_f(gi) * tanh_f(gj);
            c1s[r * HSZ + n] = c;
            float h = tanh_f(c) * sigmoid_f(go);
            x2[r * 512 + n] = (_Float16)h;           // layer2 input (K 0..255)
            gates[r * G4 + n]       = b2[n];
            gates[r * G4 + n + 256] = b2[n + 256];
            gates[r * G4 + n + 512] = b2[n + 512];
            gates[r * G4 + n + 768] = b2[n + 768];
        }
        __syncthreads();

        // ---- GEMM2: gates += [h1_t, h2_{t-1}] @ W2  (WMMA f16, K=512) ----
        #pragma unroll 1
        for (int nti = 0; nti < 16; ++nti) {
            int nt = wave + nti * 4;
            int ncol = nt * 16 + m;
            v8f c;
            #pragma unroll
            for (int v = 0; v < 8; ++v) c[v] = gates[(crow + v) * G4 + ncol];
            #pragma unroll
            for (int kt = 0; kt < 16; ++kt) {
                int koff = kt * 32 + aoff;
                v8h alo = *(const v8h*)&x2[m * 512 + koff];
                v8h ahi = *(const v8h*)&x2[m * 512 + koff + 8];
                v16h a = __builtin_shufflevector(alo, ahi,
                           0,1,2,3,4,5,6,7,8,9,10,11,12,13,14,15);
                v16h b = *(const v16h*)(w2s + (((nt * 16 + kt) * 32) + lane) * 16);
                c = __builtin_amdgcn_wmma_f32_16x16x32_f16(
                        false, a, false, b, (short)0, c, false, false);
            }
            #pragma unroll
            for (int v = 0; v < 8; ++v) gates[(crow + v) * G4 + ncol] = c[v];
        }
        __syncthreads();

        // ---- Cell 2 (+ prefetch next features) ----
        #pragma unroll 1
        for (int i = 0; i < 32; ++i) {
            int r = i >> 1;
            int n = ((i & 1) << 7) | tid;
            float gi = gates[r * G4 + n];
            float gj = gates[r * G4 + n + 256];
            float gf = gates[r * G4 + n + 512];
            float go = gates[r * G4 + n + 768];
            float c  = c2s[r * HSZ + n];
            c = c * sigmoid_f(gf + FORGET_BIAS) + sigmoid_f(gi) * tanh_f(gj);
            c2s[r * HSZ + n] = c;
            float h = tanh_f(c) * sigmoid_f(go);
            x2[r * 512 + 256 + n] = (_Float16)h;     // h2 recurrence (K 256..511)
            if (t == TSTEPS - 1) h2out[(r0 + r) * HSZ + n] = h;
        }
        if (t + 1 < TSTEPS && tid < 16)
            feats[tid] = features[(r0 + tid) * TSTEPS + (t + 1)];
        __syncthreads();
    }
}

// ---------------------------------------------------------------------------
// Per-row logits + logsumexp cross-entropy. Grid: B blocks x 128 threads.
// ---------------------------------------------------------------------------
__global__ void logits_loss_kernel(const float* __restrict__ h2,   // [B][256]
                                   const float* __restrict__ Wd,   // [256][80]
                                   const float* __restrict__ bd,   // [80]
                                   const int*   __restrict__ labels,
                                   float* __restrict__ row_loss)   // [B]
{
    __shared__ float hrow[HSZ];
    __shared__ float red[128];
    __shared__ float lgs[VOCAB];
    int b = blockIdx.x, tid = threadIdx.x;

    for (int i = tid; i < HSZ; i += 128) hrow[i] = h2[b * HSZ + i];
    __syncthreads();

    bool active = tid < VOCAB;
    float lg = 0.0f;
    if (active) {
        float acc = bd[tid];
        #pragma unroll 4
        for (int k = 0; k < HSZ; ++k) acc += hrow[k] * Wd[k * VOCAB + tid];
        lg = acc;
        lgs[tid] = acc;
    }
    red[tid] = active ? lg : -3.0e38f;
    __syncthreads();
    for (int s = 64; s > 0; s >>= 1) {
        if (tid < s) red[tid] = fmaxf(red[tid], red[tid + s]);
        __syncthreads();
    }
    float mx = red[0];
    __syncthreads();
    red[tid] = active ? __expf(lg - mx) : 0.0f;
    __syncthreads();
    for (int s = 64; s > 0; s >>= 1) {
        if (tid < s) red[tid] += red[tid + s];
        __syncthreads();
    }
    if (tid == 0) {
        float lse = __logf(red[0]) + mx;
        row_loss[b] = lse - lgs[labels[b]];
    }
}

// Deterministic tree reduction of the B row losses -> mean.
__global__ void reduce_loss_kernel(const float* __restrict__ row_loss,
                                   float* __restrict__ out)
{
    __shared__ float red[256];
    int tid = threadIdx.x;
    float s = 0.0f;
    for (int i = tid; i < BATCH; i += 256) s += row_loss[i];
    red[tid] = s;
    __syncthreads();
    for (int st = 128; st > 0; st >>= 1) {
        if (tid < st) red[tid] += red[tid + st];
        __syncthreads();
    }
    if (tid == 0) out[0] = red[0] / (float)BATCH;
}

// ---------------------------------------------------------------------------
extern "C" void kernel_launch(void* const* d_in, const int* in_sizes, int n_in,
                              void* d_out, int out_size, void* d_ws, size_t ws_size,
                              hipStream_t stream)
{
    const int*   features = (const int*)  d_in[0];
    const int*   labels   = (const int*)  d_in[1];
    const float* embedding= (const float*)d_in[2];
    const float* W1       = (const float*)d_in[3];
    const float* b1       = (const float*)d_in[4];
    const float* W2       = (const float*)d_in[5];
    const float* b2       = (const float*)d_in[6];
    const float* Wd       = (const float*)d_in[7];
    const float* bd       = (const float*)d_in[8];
    float* out = (float*)d_out;

    char* ws = (char*)d_ws;
    _Float16* wh1s   = (_Float16*)(ws);                            // 512 KB
    _Float16* w2s    = (_Float16*)(ws + (512u << 10));             // 1 MB
    float*    h2buf  = (float*)   (ws + (512u << 10) + (1u << 20));        // 2 MB
    float*    rloss  = (float*)   (ws + (512u << 10) + (1u << 20) + BATCH * HSZ * 4);

    // 1) weight convert + swizzle (f32 -> f16 B-operand layout)
    prep_weights_kernel<<<(512 * G4 + 255) / 256, 256, 0, stream>>>(W1, W2, wh1s, w2s);

    // 2) persistent fused 2-layer LSTM over all timesteps
    lstm2_persistent_kernel<<<BATCH / 16, 128, 0, stream>>>(
        features, embedding, W1, b1, b2, wh1s, w2s, h2buf);

    // 3) logits + per-row cross entropy, then deterministic mean
    logits_loss_kernel<<<BATCH, 128, 0, stream>>>(h2buf, Wd, bd, labels, rloss);
    reduce_loss_kernel<<<1, 256, 0, stream>>>(rloss, out);
}
